// Model_31258771980511
// MI455X (gfx1250) — matrix operand order repaired and checked by reading, not compiled, not run on hardware
//
#include <hip/hip_runtime.h>
#include <hip/hip_bf16.h>

// ---------------- problem constants ----------------
#define BB   8
#define TT   243
#define JJ   17
#define CC   256
#define NBLK 6
#define MR   33048           // BB*TT*JJ
#define TJ   4131            // TT*JJ

typedef __attribute__((ext_vector_type(16))) __bf16 v16bf;
typedef __attribute__((ext_vector_type(8)))  __bf16 v8bf;
typedef __attribute__((ext_vector_type(8)))  float  v8f;

// ---------------- helpers ----------------
__device__ __forceinline__ float bf2f(unsigned short h) {
  union { unsigned int u; float f; } x;
  x.u = ((unsigned int)h) << 16;
  return x.f;
}
__device__ __forceinline__ unsigned short f2bf(float f) {
  union { float f; unsigned int u; } x; x.f = f;
  unsigned int u = x.u;
  u += 0x7fffu + ((u >> 16) & 1u);      // round-to-nearest-even
  return (unsigned short)(u >> 16);
}
__device__ __forceinline__ float gelu_f(float v) {
  return 0.5f * v * (1.0f + erff(v * 0.70710678118654752f));
}

// ---------------- fp32 -> bf16 convert ----------------
__global__ void f32_to_bf16_kernel(const float* __restrict__ in,
                                   unsigned short* __restrict__ out, int n) {
  int i = blockIdx.x * 256 + threadIdx.x;
  if (i < n) out[i] = f2bf(in[i]);
}

// ---------------- embedding: z = gelu(x @ W_emb^T) ----------------
// x: (B,2,T,J,1) fp32 ; W_emb: (C,2) ; z: (M,C) fp32. One block per row.
__global__ void embed_kernel(const float* __restrict__ x,
                             const float* __restrict__ Wemb,
                             float* __restrict__ z) {
  int m = blockIdx.x;
  int c = threadIdx.x;
  int b = m / TJ, rem = m % TJ;
  float x0 = x[((size_t)b * 2 + 0) * TJ + rem];
  float x1 = x[((size_t)b * 2 + 1) * TJ + rem];
  float v = x0 * Wemb[c * 2 + 0] + x1 * Wemb[c * 2 + 1];
  z[(size_t)m * CC + c] = gelu_f(v);
}

// ---------------- LayerNorm(C=256) -> bf16. 1 wave per row ----------------
__global__ void ln_bf16_kernel(const float* __restrict__ x,
                               const float* __restrict__ g,
                               const float* __restrict__ b,
                               unsigned short* __restrict__ out) {
  int row  = blockIdx.x * 8 + (threadIdx.x >> 5);
  int lane = threadIdx.x & 31;
  const float* xr = x + (size_t)row * CC;
  float v[8]; float s = 0.f;
#pragma unroll
  for (int i = 0; i < 8; ++i) { v[i] = xr[lane + 32 * i]; s += v[i]; }
#pragma unroll
  for (int o = 16; o > 0; o >>= 1) s += __shfl_xor(s, o, 32);
  float mean = s * (1.0f / CC);
  float var = 0.f;
#pragma unroll
  for (int i = 0; i < 8; ++i) { float d = v[i] - mean; var += d * d; }
#pragma unroll
  for (int o = 16; o > 0; o >>= 1) var += __shfl_xor(var, o, 32);
  float rinv = rsqrtf(var * (1.0f / CC) + 1e-5f);
#pragma unroll
  for (int i = 0; i < 8; ++i) {
    int c = lane + 32 * i;
    out[(size_t)row * CC + c] = f2bf((v[i] - mean) * rinv * g[c] + b[c]);
  }
}

// ---------------- WMMA GEMM: out[M,N] = A[M,K](bf16) @ W[N,K]^T + epilogue ---
// block = 256 thr (8 waves). Block tile: 32(M) x 256(N); wave tile 32x32:
// 2 M-subtiles x 2 N-subtiles = 4 WMMA D tiles (4 independent chains / k-step).
// A tile staged in LDS (zero-padded tail rows -> EXEC stays all-ones for WMMA).
template <int KDIM, bool HAS_BIAS, bool HAS_RES, bool DO_GELU, bool OUT_BF16>
__global__ __launch_bounds__(256, 2)
void gemm_wmma_kernel(const unsigned short* __restrict__ A,
                      const unsigned short* __restrict__ W,
                      const float* __restrict__ bias,
                      const float* __restrict__ res,
                      void* __restrict__ outv,
                      int Mrows, int Ncols) {
  __shared__ __align__(16) unsigned short As[32 * KDIM];
  const int tid = threadIdx.x;
  const int m0 = blockIdx.x * 32;

  for (int idx = tid * 8; idx < 32 * KDIM; idx += 256 * 8) {
    const int r = idx / KDIM;
    const int m = m0 + r;
    uint4 v = make_uint4(0u, 0u, 0u, 0u);
    if (m < Mrows) v = *(const uint4*)(A + (size_t)m * KDIM + (idx % KDIM));
    *(uint4*)(As + idx) = v;
  }
  __syncthreads();

  const int wave  = tid >> 5;
  const int lane  = tid & 31;
  const int lhalf = (lane < 16) ? 0 : 1;
  const int l15   = lane & 15;
  const int nbase = blockIdx.y * 256 + wave * 32;

  const unsigned short* wrow0 = W + (size_t)(nbase + l15) * KDIM;
  const unsigned short* wrow1 = W + (size_t)(nbase + 16 + l15) * KDIM;
  const unsigned short* arow0 = As + l15 * KDIM;         // M rows m0..m0+15
  const unsigned short* arow1 = As + (16 + l15) * KDIM;  // M rows m0+16..m0+31
  const int aseg = lhalf * 8;    // A: lanes 0-15 hold K {k..k+7, k+16..k+23}; 16-31 hold +8
  const int bseg = lhalf * 8;    // B: same K order as A

  typedef union { v16bf v; v8bf h[2]; } frag;
  v8f acc00 = {}, acc01 = {}, acc10 = {}, acc11 = {};

#pragma unroll 2
  for (int ks = 0; ks < KDIM; ks += 32) {
    frag a0, a1, b0, b1;
    a0.h[0] = *(const v8bf*)(arow0 + ks + aseg);
    a0.h[1] = *(const v8bf*)(arow0 + ks + aseg + 16);
    a1.h[0] = *(const v8bf*)(arow1 + ks + aseg);
    a1.h[1] = *(const v8bf*)(arow1 + ks + aseg + 16);
    b0.h[0] = *(const v8bf*)(wrow0 + ks + bseg);
    b0.h[1] = *(const v8bf*)(wrow0 + ks + bseg + 16);
    b1.h[0] = *(const v8bf*)(wrow1 + ks + bseg);
    b1.h[1] = *(const v8bf*)(wrow1 + ks + bseg + 16);
    acc00 = __builtin_amdgcn_wmma_f32_16x16x32_bf16(false, a0.v, false, b0.v,
                                                    (short)0, acc00, false, false);
    acc01 = __builtin_amdgcn_wmma_f32_16x16x32_bf16(false, a0.v, false, b1.v,
                                                    (short)0, acc01, false, false);
    acc10 = __builtin_amdgcn_wmma_f32_16x16x32_bf16(false, a1.v, false, b0.v,
                                                    (short)0, acc10, false, false);
    acc11 = __builtin_amdgcn_wmma_f32_16x16x32_bf16(false, a1.v, false, b1.v,
                                                    (short)0, acc11, false, false);
  }

  const int n0 = nbase + l15;
  const int n1 = nbase + 16 + l15;
  float bia0 = HAS_BIAS ? bias[n0] : 0.f;
  float bia1 = HAS_BIAS ? bias[n1] : 0.f;

#pragma unroll
  for (int mt = 0; mt < 2; ++mt) {
    // D layout: lanes 0-15 -> rows base..base+7; lanes 16-31 -> base+8..base+15
    const int mb = m0 + mt * 16 + lhalf * 8;
    v8f a0 = mt ? acc10 : acc00;
    v8f a1 = mt ? acc11 : acc01;
#pragma unroll
    for (int r = 0; r < 8; ++r) {
      int m = mb + r;
      if (m >= Mrows) break;
      float v0 = a0[r] + bia0;
      float v1 = a1[r] + bia1;
      if (HAS_RES) {
        v0 += res[(size_t)m * Ncols + n0];
        v1 += res[(size_t)m * Ncols + n1];
      }
      if (DO_GELU) { v0 = gelu_f(v0); v1 = gelu_f(v1); }
      if (OUT_BF16) {
        unsigned short* o = (unsigned short*)outv;
        o[(size_t)m * Ncols + n0] = f2bf(v0);
        o[(size_t)m * Ncols + n1] = f2bf(v1);
      } else {
        float* o = (float*)outv;
        o[(size_t)m * Ncols + n0] = v0;
        o[(size_t)m * Ncols + n1] = v1;
      }
    }
  }
}

// ---------------- spatial attention (17-wide), 1 wave per (b,t,s,h) ----------
// qkv interleaved: feature n of row m at qkv[m*768 + c*3 + {0,1,2}]
__global__ void attn_s_kernel(const unsigned short* __restrict__ qkv,
                              float* __restrict__ outp) {
  const int gw = blockIdx.x * 8 + (threadIdx.x >> 5);
  if (gw >= MR * 4) return;
  const int lane = threadIdx.x & 31;
  const int h = gw & 3;
  const int m = gw >> 2;
  const int bt = m / JJ;             // b*T + t
  const int ch = h * 32 + lane;
  float qreg = bf2f(qkv[(size_t)m * 768 + (size_t)ch * 3 + 0]);

  const int zc = (lane < JJ) ? lane : (JJ - 1);       // clamp -> uniform EXEC
  const size_t krow = ((size_t)bt * JJ + zc) * 768;
  float sc = 0.f;
#pragma unroll
  for (int d = 0; d < 32; ++d) {
    float qd = __shfl(qreg, d, 32);
    sc += qd * bf2f(qkv[krow + (size_t)(h * 32 + d) * 3 + 1]);
  }
  sc = (lane < JJ) ? sc * 0.0625f : -3.0e38f;         // scale = C^-0.5
  float mx = sc;
#pragma unroll
  for (int o = 16; o > 0; o >>= 1) mx = fmaxf(mx, __shfl_xor(mx, o, 32));
  float p = (lane < JJ) ? __expf(sc - mx) : 0.f;
  float sum = p;
#pragma unroll
  for (int o = 16; o > 0; o >>= 1) sum += __shfl_xor(sum, o, 32);
  p /= sum;
  float acc = 0.f;
  for (int z = 0; z < JJ; ++z) {
    float pz = __shfl(p, z, 32);
    acc += pz * bf2f(qkv[((size_t)bt * JJ + z) * 768 + (size_t)ch * 3 + 2]);
  }
  outp[(size_t)m * 128 + ch] = acc;
}

// ---------------- temporal attention (243-wide), 1 wave per (b,s,t,h) --------
__global__ void attn_t_kernel(const unsigned short* __restrict__ qkv,
                              float* __restrict__ outp) {
  const int gw = blockIdx.x * 8 + (threadIdx.x >> 5);
  if (gw >= MR * 4) return;
  const int lane = threadIdx.x & 31;
  const int h = gw & 3;
  const int m = gw >> 2;
  const int s = m % JJ;
  const int bt = m / JJ;
  const int b = bt / TT;
  const int ch = 128 + h * 32 + lane;
  float qreg = bf2f(qkv[(size_t)m * 768 + (size_t)ch * 3 + 0]);

  float sc[8];
  float mx = -3.0e38f;
#pragma unroll 1
  for (int j = 0; j < 8; ++j) {
    int z = lane + j * 32;
    int zcl = (z < TT) ? z : (TT - 1);                 // clamp -> uniform EXEC
    size_t krow = ((size_t)(b * TT + zcl) * JJ + s) * 768;
    float acc = 0.f;
#pragma unroll
    for (int d = 0; d < 32; ++d) {
      float qd = __shfl(qreg, d, 32);
      acc += qd * bf2f(qkv[krow + (size_t)(128 + h * 32 + d) * 3 + 1]);
    }
    acc = (z < TT) ? acc * 0.0625f : -3.0e38f;
    sc[j] = acc;
    mx = fmaxf(mx, acc);
  }
#pragma unroll
  for (int o = 16; o > 0; o >>= 1) mx = fmaxf(mx, __shfl_xor(mx, o, 32));
  float p[8]; float sum = 0.f;
#pragma unroll
  for (int j = 0; j < 8; ++j) {
    int z = lane + j * 32;
    p[j] = (z < TT) ? __expf(sc[j] - mx) : 0.f;
    sum += p[j];
  }
#pragma unroll
  for (int o = 16; o > 0; o >>= 1) sum += __shfl_xor(sum, o, 32);
  float inv = 1.0f / sum;
  float acc = 0.f;
  for (int j = 0; j < 8; ++j) {
#pragma unroll 1
    for (int zl = 0; zl < 32; ++zl) {
      int z = j * 32 + zl;
      if (z >= TT) break;                              // z lane-uniform
      float pz = __shfl(p[j], zl, 32);
      acc += pz * bf2f(qkv[((size_t)(b * TT + z) * JJ + s) * 768 +
                           (size_t)ch * 3 + 2]);
    }
  }
  outp[(size_t)m * 128 + (h * 32 + lane)] = acc * inv;
}

// ------- assemble xc: att + depthwise 3x3 gate conv + 1e-4*sep, -> bf16 ------
// xc[m, h*64+d]    = atts[m,h*32+d] + dwconv_s + 1e-4*emb[part[s], d]   (d<32)
// xc[m, h*64+32+d] = attt[m,h*32+d] + dwconv_t
__global__ void assemble_kernel(const unsigned short* __restrict__ qkv,
                                const float* __restrict__ atts,
                                const float* __restrict__ attt,
                                const float* __restrict__ gsw,
                                const float* __restrict__ gsb,
                                const float* __restrict__ gtw,
                                const float* __restrict__ gtb,
                                const float* __restrict__ emb,  // (20,32) of block
                                const int* __restrict__ part,
                                unsigned short* __restrict__ xc) {
  int m = blockIdx.x;
  int c = threadIdx.x;
  int h = c >> 6, dd = c & 63;
  int s = m % JJ, bt = m / JJ, t = bt % TT, b = bt / TT;
  bool sp = dd < 32;
  int ch = h * 32 + (sp ? dd : dd - 32);
  int qch = sp ? ch : (128 + ch);
  const float* w = (sp ? gsw : gtw) + ch * 9;
  float acc = (sp ? gsb : gtb)[ch];
#pragma unroll
  for (int di = -1; di <= 1; ++di) {
    int tt = t + di;
    if (tt < 0 || tt >= TT) continue;
#pragma unroll
    for (int dj = -1; dj <= 1; ++dj) {
      int ss = s + dj;
      if (ss < 0 || ss >= JJ) continue;
      size_t mm = ((size_t)(b * TT + tt)) * JJ + ss;
      acc += w[(di + 1) * 3 + (dj + 1)] *
             bf2f(qkv[mm * 768 + (size_t)qch * 3 + 2]);
    }
  }
  float base = sp ? atts[(size_t)m * 128 + ch] : attt[(size_t)m * 128 + ch];
  float v = base + acc;
  if (sp) v += 1e-4f * emb[part[s] * 32 + dd];
  xc[(size_t)m * CC + c] = f2bf(v);
}

// ---------------- final lift: out = z @ W_lift^T (N=3) ----------------
__global__ void lift_kernel(const float* __restrict__ z,
                            const float* __restrict__ Wl,
                            float* __restrict__ out) {
  int i = blockIdx.x * 256 + threadIdx.x;
  if (i >= MR * 3) return;
  int m = i / 3, o = i % 3;
  const float* zr = z + (size_t)m * CC;
  const float* wr = Wl + o * CC;
  float acc = 0.f;
  for (int c = 0; c < CC; ++c) acc += zr[c] * wr[c];
  out[i] = acc;
}

// ============================ host side ============================
extern "C" void kernel_launch(void* const* d_in, const int* in_sizes, int n_in,
                              void* d_out, int out_size, void* d_ws, size_t ws_size,
                              hipStream_t stream) {
  const float* x        = (const float*)d_in[0];
  const float* W_emb    = (const float*)d_in[1];
  const float* W_lift   = (const float*)d_in[2];
  const float* ln_att_g = (const float*)d_in[3];
  const float* ln_att_b = (const float*)d_in[4];
  const float* Wqkv     = (const float*)d_in[5];
  const float* bqkv     = (const float*)d_in[6];
  const float* gate_s_w = (const float*)d_in[7];
  const float* gate_s_b = (const float*)d_in[8];
  const float* gate_t_w = (const float*)d_in[9];
  const float* gate_t_b = (const float*)d_in[10];
  const float* emb_tab  = (const float*)d_in[11];
  const float* Wproj    = (const float*)d_in[12];
  const float* bproj    = (const float*)d_in[13];
  const float* ln1_g    = (const float*)d_in[14];
  const float* ln1_b    = (const float*)d_in[15];
  const float* Wfc1     = (const float*)d_in[16];
  const float* Wfc2     = (const float*)d_in[17];
  const int*   part     = (const int*)d_in[18];
  float* out = (float*)d_out;

  size_t off = 0;
  char* base = (char*)d_ws;
  auto carve = [&](size_t bytes) -> void* {
    void* r = base + off;
    off += (bytes + 255) & ~(size_t)255;
    return r;
  };
  float*          z     = (float*)carve((size_t)MR * 256 * 4);
  unsigned short* zb    = (unsigned short*)carve((size_t)MR * 256 * 2);
  unsigned short* qkvb  = (unsigned short*)carve((size_t)MR * 768 * 2);
  float*          atts  = (float*)carve((size_t)MR * 128 * 4);
  float*          attt  = (float*)carve((size_t)MR * 128 * 4);
  unsigned short* xcb   = (unsigned short*)carve((size_t)MR * 256 * 2);
  unsigned short* hb    = (unsigned short*)carve((size_t)MR * 1024 * 2);
  unsigned short* WqkvB = (unsigned short*)carve((size_t)NBLK * 768 * 256 * 2);
  unsigned short* WprojB= (unsigned short*)carve((size_t)NBLK * 256 * 256 * 2);
  unsigned short* Wfc1B = (unsigned short*)carve((size_t)NBLK * 1024 * 256 * 2);
  unsigned short* Wfc2B = (unsigned short*)carve((size_t)NBLK * 1024 * 256 * 2);

  // ---- weights -> bf16 (once per launch; deterministic) ----
  {
    int n;
    n = NBLK * 768 * 256;
    f32_to_bf16_kernel<<<(n + 255) / 256, 256, 0, stream>>>(Wqkv, WqkvB, n);
    n = NBLK * 256 * 256;
    f32_to_bf16_kernel<<<(n + 255) / 256, 256, 0, stream>>>(Wproj, WprojB, n);
    n = NBLK * 1024 * 256;
    f32_to_bf16_kernel<<<(n + 255) / 256, 256, 0, stream>>>(Wfc1, Wfc1B, n);
    n = NBLK * 1024 * 256;
    f32_to_bf16_kernel<<<(n + 255) / 256, 256, 0, stream>>>(Wfc2, Wfc2B, n);
  }

  // ---- embedding ----
  embed_kernel<<<MR, 256, 0, stream>>>(x, W_emb, z);

  const int gm = (MR + 31) / 32;          // 1033 M-tiles (32 rows each)
  const dim3 blk256(256);

  for (int i = 0; i < NBLK; ++i) {
    // LN(att) -> bf16
    ln_bf16_kernel<<<MR / 8, 256, 0, stream>>>(z, ln_att_g + i * 256,
                                               ln_att_b + i * 256, zb);
    // QKV = zb @ Wqkv^T + bqkv   (M x 768, bf16 out)
    gemm_wmma_kernel<256, true, false, false, true>
        <<<dim3(gm, 768 / 256), blk256, 0, stream>>>(
            zb, WqkvB + (size_t)i * 768 * 256, bqkv + i * 768, nullptr,
            (void*)qkvb, MR, 768);
    // attention
    attn_s_kernel<<<MR * 4 / 8, 256, 0, stream>>>(qkvb, atts);
    attn_t_kernel<<<MR * 4 / 8, 256, 0, stream>>>(qkvb, attt);
    // gates + sep + concat -> xc (bf16)
    assemble_kernel<<<MR, 256, 0, stream>>>(
        qkvb, atts, attt,
        gate_s_w + (size_t)i * 128 * 9, gate_s_b + i * 128,
        gate_t_w + (size_t)i * 128 * 9, gate_t_b + i * 128,
        emb_tab + (size_t)i * 20 * 32, part, xcb);
    // z = xc @ Wproj^T + bproj + z   (fp32, fused residual)
    gemm_wmma_kernel<256, true, true, false, false>
        <<<dim3(gm, 256 / 256), blk256, 0, stream>>>(
            xcb, WprojB + (size_t)i * 256 * 256, bproj + i * 256, z,
            (void*)z, MR, 256);
    // LN1 -> bf16
    ln_bf16_kernel<<<MR / 8, 256, 0, stream>>>(z, ln1_g + i * 256,
                                               ln1_b + i * 256, zb);
    // h = gelu(zb @ Wfc1^T)  (bf16 out, no bias)
    gemm_wmma_kernel<256, false, false, true, true>
        <<<dim3(gm, 1024 / 256), blk256, 0, stream>>>(
            zb, Wfc1B + (size_t)i * 1024 * 256, nullptr, nullptr,
            (void*)hb, MR, 1024);
    // z = h @ Wfc2^T + z   (fp32, fused residual, K=1024)
    gemm_wmma_kernel<1024, false, true, false, false>
        <<<dim3(gm, 256 / 256), blk256, 0, stream>>>(
            hb, Wfc2B + (size_t)i * 1024 * 256, nullptr, z,
            (void*)z, MR, 256);
  }

  // ---- lift ----
  lift_kernel<<<(MR * 3 + 255) / 256, 256, 0, stream>>>(z, W_lift, out);
}
